// ImageMoELayer_36842229465896
// MI455X (gfx1250) — matrix operand order repaired and checked by reading, not compiled, not run on hardware
//
#include <hip/hip_runtime.h>
#include <hip/hip_bf16.h>

#define T_TOK  32768
#define HID    1024
#define INTER  2048
#define NEXP   8
#define EPSF   1e-5f

#if defined(__has_builtin)
#if __has_builtin(__builtin_amdgcn_global_load_async_to_lds_b128)
#define HAVE_ASYNC_LDS 1
#endif
#endif
#ifndef HAVE_ASYNC_LDS
#define HAVE_ASYNC_LDS 0
#endif

typedef __attribute__((ext_vector_type(16))) __bf16 v16bf;
typedef __attribute__((ext_vector_type(8)))  __bf16 v8bf;
typedef __attribute__((ext_vector_type(8)))  float  v8f;
typedef int v4i __attribute__((vector_size(16)));   // matches builtin param type

__device__ __forceinline__ unsigned short f2bfbits(float f) {
    unsigned int u = __builtin_bit_cast(unsigned int, f);
    return (unsigned short)((u + 0x7FFFu + ((u >> 16) & 1u)) >> 16);
}
__device__ __forceinline__ __bf16 f2bf(float f) {
    unsigned short s = f2bfbits(f);
    return __builtin_bit_cast(__bf16, s);
}

__device__ __forceinline__ float fast_rcp(float x) {
#if defined(__has_builtin)
#if __has_builtin(__builtin_amdgcn_rcpf)
    return __builtin_amdgcn_rcpf(x);
#else
    return 1.f / x;
#endif
#else
    return 1.f / x;
#endif
}

__device__ __forceinline__ float wave_sum(float v) {
    #pragma unroll
    for (int off = 16; off > 0; off >>= 1) v += __shfl_xor(v, off, 32);
    return v;
}

__device__ __forceinline__ v8f wmma_bf16(v16bf a, v16bf b, v8f c) {
    return __builtin_amdgcn_wmma_f32_16x16x32_bf16(false, a, false, b, (short)0, c, false, false);
}

__device__ __forceinline__ void wait_async0() {
#if HAVE_ASYNC_LDS
#if __has_builtin(__builtin_amdgcn_s_wait_asynccnt)
    __builtin_amdgcn_s_wait_asynccnt(0);
#else
    asm volatile("s_wait_asynccnt 0x0" ::: "memory");
#endif
#endif
}

// ---------------------------------------------------------------- router ----
__global__ __launch_bounds__(256) void init_cnt_kernel(int* cnt) {
    if (threadIdx.x < NEXP) cnt[threadIdx.x] = 0;
}

__global__ __launch_bounds__(256) void router_kernel(
    const float* __restrict__ x, const float* __restrict__ ln_s,
    const float* __restrict__ ln_b, const float* __restrict__ gate_w,
    float* __restrict__ wout, int* __restrict__ cnt, int* __restrict__ idx)
{
    int wave = threadIdx.x >> 5;
    int lane = threadIdx.x & 31;
    int t = blockIdx.x * 8 + wave;
    const float* xr = x + (size_t)t * HID;

    float xv[32];
    #pragma unroll
    for (int i = 0; i < 32; ++i) xv[i] = xr[lane + 32 * i];

    float s = 0.f, s2 = 0.f;
    #pragma unroll
    for (int i = 0; i < 32; ++i) { s += xv[i]; s2 += xv[i] * xv[i]; }
    s = wave_sum(s); s2 = wave_sum(s2);
    float mu = s * (1.f / HID);
    float var = s2 * (1.f / HID) - mu * mu;
    float rs = rsqrtf(var + EPSF);

    float xn[32];
    #pragma unroll
    for (int i = 0; i < 32; ++i) {
        int j = lane + 32 * i;
        xn[i] = (xv[i] - mu) * rs * ln_s[j] + ln_b[j];
    }

    float logits[NEXP];
    #pragma unroll
    for (int e = 0; e < NEXP; ++e) {
        const float* gr = gate_w + (size_t)e * HID;
        float a = 0.f;
        #pragma unroll
        for (int i = 0; i < 32; ++i) a += xn[i] * gr[lane + 32 * i];
        logits[e] = wave_sum(a);   // broadcast to all lanes
    }

    float mx = logits[0];
    #pragma unroll
    for (int e = 1; e < NEXP; ++e) mx = fmaxf(mx, logits[e]);
    float pe[NEXP]; float ssum = 0.f;
    #pragma unroll
    for (int e = 0; e < NEXP; ++e) { pe[e] = __expf(logits[e] - mx); ssum += pe[e]; }
    float inv = 1.f / ssum;
    #pragma unroll
    for (int e = 0; e < NEXP; ++e) pe[e] *= inv;

    int e1 = 0; float p1 = pe[0];
    #pragma unroll
    for (int e = 1; e < NEXP; ++e) if (pe[e] > p1) { p1 = pe[e]; e1 = e; }
    int e2 = -1; float p2 = -1.f;
    #pragma unroll
    for (int e = 0; e < NEXP; ++e) if (e != e1 && pe[e] > p2) { p2 = pe[e]; e2 = e; }
    float r = 1.f / (p1 + p2 + EPSF);
    float w1 = p1 * r, w2 = p2 * r;

    if (lane < NEXP)
        wout[(size_t)t * NEXP + lane] = (lane == e1) ? w1 : ((lane == e2) ? w2 : 0.f);
    if (lane == 0) {
        int p = atomicAdd(&cnt[e1], 1); idx[e1 * T_TOK + p] = t;
        int q = atomicAdd(&cnt[e2], 1); idx[e2 * T_TOK + q] = t;
    }
}

// ----------------------------------------------------------- conversions ----
__global__ __launch_bounds__(256) void cvt_x_kernel(
    const float4* __restrict__ in, ushort4* __restrict__ outb, int n4)
{
    int i = blockIdx.x * 256 + threadIdx.x;
    if (i < n4) {
        float4 v = in[i];
        outb[i] = make_ushort4(f2bfbits(v.x), f2bfbits(v.y), f2bfbits(v.z), f2bfbits(v.w));
    }
}

// src slice [R,C] fp32 -> dst slice [C,R] bf16, slice selected by blockIdx.z
__global__ __launch_bounds__(256) void transpose_cvt_kernel(
    const float* __restrict__ src, __bf16* __restrict__ dst, int R, int C)
{
    __shared__ float tile[32][33];
    size_t slice = (size_t)blockIdx.z * R * C;
    src += slice; dst += slice;
    int r0 = blockIdx.y * 32, c0 = blockIdx.x * 32;
    int tx = threadIdx.x, ty = threadIdx.y;     // 32 x 8
    #pragma unroll
    for (int i = 0; i < 32; i += 8)
        tile[ty + i][tx] = src[(size_t)(r0 + ty + i) * C + (c0 + tx)];
    __syncthreads();
    #pragma unroll
    for (int i = 0; i < 32; i += 8)
        dst[(size_t)(c0 + ty + i) * R + (r0 + tx)] = f2bf(tile[tx][ty + i]);
}

// ------------------------------------------------------------ GEMM pieces ---
// Stage A tile [128 rows x 64 K] bf16 into LDS (optional row gather).
// Async path: GLOBAL_LOAD_ASYNC_TO_LDS_B128 (ASYNCcnt); fallback: VGPR copy.
__device__ __forceinline__ void stage_A(const __bf16* __restrict__ Asrc, int lda,
                                        const int* __restrict__ gidx, int cnt,
                                        int mbase, int kb, __bf16* As)
{
    int th = threadIdx.x;             // 256 threads
    int row = th >> 1, half = th & 1; // 2 threads/row, 64B each
    int slot = mbase + row;
    int t = gidx ? (slot < cnt ? gidx[slot] : 0) : slot;
    const __bf16* gp = Asrc + (size_t)t * lda + kb + half * 32;
    __bf16* lp = As + row * 64 + half * 32;
#if HAVE_ASYNC_LDS
    v4i* gq = (v4i*)gp;   // generic -> builtin's global AS via implicit conversion
    v4i* lq = (v4i*)lp;   // generic -> builtin's LDS AS via implicit conversion
    __builtin_amdgcn_global_load_async_to_lds_b128(gq, lq, 0, 0);
    __builtin_amdgcn_global_load_async_to_lds_b128(gq, lq, 16, 0);
    __builtin_amdgcn_global_load_async_to_lds_b128(gq, lq, 32, 0);
    __builtin_amdgcn_global_load_async_to_lds_b128(gq, lq, 48, 0);
#else
    const uint4* g4 = (const uint4*)gp;
    uint4* l4 = (uint4*)lp;
    #pragma unroll
    for (int i = 0; i < 4; ++i) l4[i] = g4[i];
#endif
}

// A fragment (16x32 bf16): lane<16 -> K 0..15, lane>=16 -> K 16..31
__device__ __forceinline__ v16bf load_A_frag(const __bf16* As, int mrow, int kk, int lane) {
    int m = mrow + (lane & 15);
    int kh = kk + (lane >> 4) * 16;
    const v8bf* lo = (const v8bf*)(As + m * 64 + kh);
    const v8bf* hi = (const v8bf*)(As + m * 64 + 8 + kh);
    v8bf a0 = *lo, a1 = *hi;
    v16bf a;
    #pragma unroll
    for (int i = 0; i < 8; ++i) { a[i] = a0[i]; a[i + 8] = a1[i]; }
    return a;
}

// B fragment (32x16 bf16) from transposed weight W^T [N,K]: lane<16 -> K 0..15, lane>=16 -> K 16..31
__device__ __forceinline__ v16bf load_B_frag(const __bf16* __restrict__ WT, int ldb,
                                             int nb, int kb, int lane) {
    int n = nb + (lane & 15);
    int ks = kb + (lane >> 4) * 16;
    return *(const v16bf*)(WT + (size_t)n * ldb + ks);
}

// h[slot, nb..] = silu(x@Wg) * (x@Wu), bf16.  Block tile 128x128, K = HID.
__global__ __launch_bounds__(256) void gemm_gateup_kernel(
    const __bf16* __restrict__ xb, const __bf16* __restrict__ WgT,
    const __bf16* __restrict__ WuT, __bf16* __restrict__ hout,
    const int* __restrict__ gidx, const int* __restrict__ cntPtr, int Ttot)
{
    __shared__ __bf16 As[2][128 * 64];
    int cnt = cntPtr ? *cntPtr : Ttot;
    int mbase = blockIdx.x * 128;
    if (mbase >= cnt) return;
    int nbase = blockIdx.y * 128;
    int wid = threadIdx.x >> 5, lane = threadIdx.x & 31;
    int wm = (wid >> 2) * 64;   // 2 waves in M
    int wn = (wid & 3) * 32;    // 4 waves in N

    v8f cg[4][2] = {};
    v8f cu[4][2] = {};

    stage_A(xb, HID, gidx, cnt, mbase, 0, As[0]);

    for (int kb = 0; kb < HID; kb += 64) {
        int cur = (kb >> 6) & 1, nxt = cur ^ 1;
        wait_async0();
        __syncthreads();                      // As[cur] ready; all reads of As[nxt] done
        if (kb + 64 < HID)
            stage_A(xb, HID, gidx, cnt, mbase, kb + 64, As[nxt]);

        #pragma unroll
        for (int kk = 0; kk < 64; kk += 32) {
            v16bf bg[2], bu[2];
            #pragma unroll
            for (int ni = 0; ni < 2; ++ni) {
                bg[ni] = load_B_frag(WgT, HID, nbase + wn + ni * 16, kb + kk, lane);
                bu[ni] = load_B_frag(WuT, HID, nbase + wn + ni * 16, kb + kk, lane);
            }
            #pragma unroll
            for (int mi = 0; mi < 4; ++mi) {
                v16bf a = load_A_frag(As[cur], wm + mi * 16, kk, lane);
                #pragma unroll
                for (int ni = 0; ni < 2; ++ni) {
                    cg[mi][ni] = wmma_bf16(a, bg[ni], cg[mi][ni]);
                    cu[mi][ni] = wmma_bf16(a, bu[ni], cu[mi][ni]);
                }
            }
        }
    }

    int rsel = (lane >> 4) * 8;
    int ncol = lane & 15;
    #pragma unroll
    for (int mi = 0; mi < 4; ++mi) {
        #pragma unroll
        for (int ni = 0; ni < 2; ++ni) {
            #pragma unroll
            for (int r = 0; r < 8; ++r) {
                int slot = mbase + wm + mi * 16 + rsel + r;
                if (slot >= cnt) continue;
                int n = nbase + wn + ni * 16 + ncol;
                float g = cg[mi][ni][r], u = cu[mi][ni][r];
                float sg = g * fast_rcp(1.f + __expf(-g));
                hout[(size_t)slot * INTER + n] = f2bf(sg * u);
            }
        }
    }
}

// out[idx[slot], nb..] (+)= w * (h @ Wd).  Block tile 128x128, K = INTER.
__global__ __launch_bounds__(256) void gemm_down_kernel(
    const __bf16* __restrict__ hsrc, const __bf16* __restrict__ WdT,
    float* __restrict__ out, const int* __restrict__ gidx,
    const int* __restrict__ cntPtr, int Ttot,
    const float* __restrict__ wsc, int accumulate)
{
    __shared__ __bf16 As[2][128 * 64];
    int cnt = cntPtr ? *cntPtr : Ttot;
    int mbase = blockIdx.x * 128;
    if (mbase >= cnt) return;
    int nbase = blockIdx.y * 128;
    int wid = threadIdx.x >> 5, lane = threadIdx.x & 31;
    int wm = (wid >> 2) * 64;
    int wn = (wid & 3) * 32;

    v8f c[4][2] = {};

    stage_A(hsrc, INTER, nullptr, cnt, mbase, 0, As[0]);

    for (int kb = 0; kb < INTER; kb += 64) {
        int cur = (kb >> 6) & 1, nxt = cur ^ 1;
        wait_async0();
        __syncthreads();
        if (kb + 64 < INTER)
            stage_A(hsrc, INTER, nullptr, cnt, mbase, kb + 64, As[nxt]);

        #pragma unroll
        for (int kk = 0; kk < 64; kk += 32) {
            v16bf b[2];
            #pragma unroll
            for (int ni = 0; ni < 2; ++ni)
                b[ni] = load_B_frag(WdT, INTER, nbase + wn + ni * 16, kb + kk, lane);
            #pragma unroll
            for (int mi = 0; mi < 4; ++mi) {
                v16bf a = load_A_frag(As[cur], wm + mi * 16, kk, lane);
                #pragma unroll
                for (int ni = 0; ni < 2; ++ni)
                    c[mi][ni] = wmma_bf16(a, b[ni], c[mi][ni]);
            }
        }
    }

    int rsel = (lane >> 4) * 8;
    int ncol = lane & 15;
    #pragma unroll
    for (int mi = 0; mi < 4; ++mi) {
        #pragma unroll
        for (int ni = 0; ni < 2; ++ni) {
            #pragma unroll
            for (int r = 0; r < 8; ++r) {
                int slot = mbase + wm + mi * 16 + rsel + r;
                if (slot >= cnt) continue;
                int t = gidx ? gidx[slot] : slot;
                int n = nbase + wn + ni * 16 + ncol;
                float v = c[mi][ni][r];
                if (wsc) v *= wsc[(size_t)t * NEXP];
                float* op = out + (size_t)t * HID + n;
                if (accumulate) *op += v; else *op = v;
            }
        }
    }
}

// ------------------------------------------------------------------ launch --
extern "C" void kernel_launch(void* const* d_in, const int* in_sizes, int n_in,
                              void* d_out, int out_size, void* d_ws, size_t ws_size,
                              hipStream_t stream) {
    const float* x    = (const float*)d_in[0];
    const float* ln_s = (const float*)d_in[1];
    const float* ln_b = (const float*)d_in[2];
    const float* gate = (const float*)d_in[3];
    const float* Wg   = (const float*)d_in[4];
    const float* Wu   = (const float*)d_in[5];
    const float* Wd   = (const float*)d_in[6];
    const float* sWg  = (const float*)d_in[7];
    const float* sWu  = (const float*)d_in[8];
    const float* sWd  = (const float*)d_in[9];
    float* out = (float*)d_out;

    const size_t T = T_TOK, H = HID, I = INTER, E = NEXP;
    char* ws = (char*)d_ws;
    size_t off = 0;
    auto alloc = [&](size_t bytes) {
        size_t o = off; off += (bytes + 255) & ~(size_t)255; return o;
    };
    __bf16* xb    = (__bf16*)(ws + alloc(T * H * 2));
    float*  wbuf  = (float*) (ws + alloc(T * E * 4));
    int*    cnt   = (int*)   (ws + alloc(E * 4));
    int*    idx   = (int*)   (ws + alloc(E * T * 4));
    __bf16* WgT   = (__bf16*)(ws + alloc(E * H * I * 2));
    __bf16* WuT   = (__bf16*)(ws + alloc(E * H * I * 2));
    __bf16* WdT   = (__bf16*)(ws + alloc(E * H * I * 2));
    __bf16* sWgT  = (__bf16*)(ws + alloc(H * I * 2));
    __bf16* sWuT  = (__bf16*)(ws + alloc(H * I * 2));
    __bf16* sWdT  = (__bf16*)(ws + alloc(H * I * 2));
    __bf16* hbuf  = (__bf16*)(ws + alloc(T * I * 2));
    (void)ws_size; (void)in_sizes; (void)n_in; (void)out_size;

    init_cnt_kernel<<<1, 256, 0, stream>>>(cnt);
    router_kernel<<<T_TOK / 8, 256, 0, stream>>>(x, ln_s, ln_b, gate, wbuf, cnt, idx);

    int n4 = (int)(T * H / 4);
    cvt_x_kernel<<<(n4 + 255) / 256, 256, 0, stream>>>((const float4*)x, (ushort4*)xb, n4);

    dim3 tb(32, 8);
    transpose_cvt_kernel<<<dim3(I / 32, H / 32, E), tb, 0, stream>>>(Wg,  WgT,  (int)H, (int)I);
    transpose_cvt_kernel<<<dim3(I / 32, H / 32, E), tb, 0, stream>>>(Wu,  WuT,  (int)H, (int)I);
    transpose_cvt_kernel<<<dim3(H / 32, I / 32, E), tb, 0, stream>>>(Wd,  WdT,  (int)I, (int)H);
    transpose_cvt_kernel<<<dim3(I / 32, H / 32, 1), tb, 0, stream>>>(sWg, sWgT, (int)H, (int)I);
    transpose_cvt_kernel<<<dim3(I / 32, H / 32, 1), tb, 0, stream>>>(sWu, sWuT, (int)H, (int)I);
    transpose_cvt_kernel<<<dim3(H / 32, I / 32, 1), tb, 0, stream>>>(sWd, sWdT, (int)I, (int)H);

    dim3 g1(T_TOK / 128, INTER / 128);
    dim3 g2(T_TOK / 128, HID / 128);

    // Shared expert first: initializes d_out (store mode, all tokens).
    gemm_gateup_kernel<<<g1, 256, 0, stream>>>(xb, sWgT, sWuT, hbuf, nullptr, nullptr, T_TOK);
    gemm_down_kernel<<<g2, 256, 0, stream>>>(hbuf, sWdT, out, nullptr, nullptr, T_TOK, nullptr, 0);

    // Routed experts: only their gathered tokens (top-2 sparsity), accumulate.
    for (int e = 0; e < NEXP; ++e) {
        size_t wo = (size_t)e * H * I;
        gemm_gateup_kernel<<<g1, 256, 0, stream>>>(xb, WgT + wo, WuT + wo, hbuf,
                                                   idx + e * T_TOK, cnt + e, T_TOK);
        gemm_down_kernel<<<g2, 256, 0, stream>>>(hbuf, WdT + wo, out,
                                                 idx + e * T_TOK, cnt + e, T_TOK,
                                                 wbuf + e, 1);
    }
}
